// EqMotion_34488587387562
// MI455X (gfx1250) — matrix operand order repaired and checked by reading, not compiled, not run on hardware
//
#include <hip/hip_runtime.h>
#include <hip/hip_bf16.h>
#include <math.h>
#include <stdint.h>

// ---------------------------------------------------------------------------
// EqMotion forward, fused for MI455X (gfx1250, wave32, WMMA 16x16x32 f16).
// Edge MLP chains run as per-wave WMMA GEMMs with LDS-staged activations and
// pre-transposed f16 weights ([out][in], contiguous 32B B-fragments per lane).
// ---------------------------------------------------------------------------

typedef _Float16 h16;
typedef __attribute__((ext_vector_type(16))) _Float16 v16h;
typedef __attribute__((ext_vector_type(8)))  _Float16 v8h;
typedef __attribute__((ext_vector_type(8)))  float    v8f;

#define DEV __device__ __forceinline__

constexpr int B = 8, N = 128, T = 10, C = 64, H = 64, KCAT = 2, L = 4, OUTT = 20;
constexpr int TWOC = 2 * C;              // 128
constexpr float TAO = 0.5f;
constexpr float EPSF = 1e-6f;
constexpr float PI_F = 3.14159265358979323846f;
constexpr int NW = 4;                    // waves per block in edge kernels
constexpr int NTILE = N / 16;            // 8 n-tiles per node

DEV float siluf(float x) { return x * (1.0f / (1.0f + __expf(-x))); }

// ---------------------------------------------------------------------------
// 16-row tile GEMM: D[16 x OUTF] = A[16 x INF] * W[INF x OUTF], via
// v_wmma_f32_16x16x32_f16. A in LDS (row-major f16, stride lda, 16B-aligned
// rows). WT in global, f16, transposed [OUTF][INF] so each lane's B fragment
// is 32 contiguous bytes. Epilogue lambda gets (ntile, r, row, col, value).
// Wave-uniform; EXEC all ones required (call sites are non-divergent).
// ---------------------------------------------------------------------------
template <int INF, int OUTF, typename EPI>
DEV void gemm16(const h16* __restrict__ A, int lda, const h16* __restrict__ WT, EPI&& epi) {
  constexpr int NT = OUTF / 16;
  constexpr int KC = INF / 32;
  const int l = threadIdx.x & 31;
  const int mn = l & 15;       // A-row / B-col within lane group
  const int hi = l >> 4;       // lane group 0/1
  v8f acc[NT];
#pragma unroll
  for (int t = 0; t < NT; t++) {
#pragma unroll
    for (int i = 0; i < 8; i++) acc[t][i] = 0.0f;
  }
#pragma unroll
  for (int kc = 0; kc < KC; kc++) {
    const int kb = kc * 32;
    // A operand, contiguous k order as B: lanes 0-15 hold K = kb+{0..15},
    // lanes 16-31 hold K = kb+{16..31}. Two 16B contiguous runs.
    const h16* ap = A + mn * lda + kb + hi * 16;
    v8h a0 = *(const v8h*)(ap);
    v8h a1 = *(const v8h*)(ap + 8);
    v16h a;
#pragma unroll
    for (int i = 0; i < 8; i++) { a[i] = a0[i]; a[i + 8] = a1[i]; }
#pragma unroll
    for (int t = 0; t < NT; t++) {
      // B operand, contiguous k order as A: lanes 0-15 K=kb..kb+15, lanes 16-31 K=kb+16..31,
      // N = lane&15. Contiguous 16 halves per lane from transposed weights.
      const h16* bp = WT + (size_t)(t * 16 + mn) * INF + kb + hi * 16;
      v16h bf = *(const v16h*)(bp);
      acc[t] = __builtin_amdgcn_wmma_f32_16x16x32_f16(false, a, false, bf, (short)0,
                                                      acc[t], false, false);
    }
  }
  // D layout: VGPR r -> rows (r, r+8) for lane groups (0,1); col = t*16 + (lane&15)
#pragma unroll
  for (int t = 0; t < NT; t++) {
#pragma unroll
    for (int r = 0; r < 8; r++) {
      epi(t, r, r + hi * 8, t * 16 + mn, acc[t][r]);
    }
  }
}

// ---------------------------------------------------------------------------
// Weight conversion: src f32 [in][out] -> dst f16 transposed [out][in]
// ---------------------------------------------------------------------------
__global__ void k_transpose_f16(const float* __restrict__ src, h16* __restrict__ dst,
                                int in, int out) {
  int i = blockIdx.x * blockDim.x + threadIdx.x;
  if (i >= in * out) return;
  int o = i / in, k = i % in;
  dst[(size_t)o * in + k] = (h16)src[(size_t)k * out + o];
}

// mean over n of x[b,n,t,d] -> M[b,t,d]
__global__ void k_mean_nt(const float* __restrict__ x, float* __restrict__ M) {
  __shared__ float red[128];
  int b = blockIdx.x / (T * 3);
  int r = blockIdx.x % (T * 3);
  int t = r / 3, d = r % 3;
  int n = threadIdx.x;
  red[n] = x[(((size_t)b * N + n) * T + t) * 3 + d];
  __syncthreads();
  for (int s = 64; s > 0; s >>= 1) {
    if (n < s) red[n] += red[n + s];
    __syncthreads();
  }
  if (n == 0) M[(b * T + t) * 3 + d] = red[0] * (1.0f / N);
}

// x_center[b,d] and x_mean[b,d] (mean of DCT'd centered coords over n,k)
__global__ void k_centers(const float* __restrict__ M, float* __restrict__ xcen,
                          float* __restrict__ xmean) {
  int i = threadIdx.x;
  if (i >= B * 3) return;
  int b = i / 3, d = i % 3;
  float cs[T];
  float xc = 0.0f;
  for (int t = 0; t < T; t++) {
    float s = 0.0f;
    for (int k = 0; k < T; k++) {
      float w = (k == 0) ? sqrtf(1.0f / T) : sqrtf(2.0f / T);
      s += w * cosf(PI_F * (t + 0.5f) * k / T);
    }
    cs[t] = s;
    xc += M[(b * T + t) * 3 + d];
  }
  xc *= (1.0f / T);
  float xm = 0.0f;
  for (int t = 0; t < T; t++) xm += cs[t] * (M[(b * T + t) * 3 + d] - xc);
  xm *= (1.0f / T);
  xcen[b * 3 + d] = xc;
  xmean[b * 3 + d] = xm;
}

// per-node prep: vel angle, embeddings, DCT, coord/vel transforms to C-space
__global__ __launch_bounds__(128) void k_prep(
    const float* __restrict__ h_in, const float* __restrict__ x_in,
    const float* __restrict__ v_in, const float* __restrict__ emb_w,
    const float* __restrict__ emb_b, const float* __restrict__ emb2_w,
    const float* __restrict__ emb2_b, const float* __restrict__ ctw,
    const float* __restrict__ vtw, const float* __restrict__ xcen,
    const float* __restrict__ xmean, float* __restrict__ hh,
    float* __restrict__ xc, float* __restrict__ vc) {
  int b = blockIdx.x, n = threadIdx.x;
  const float* xp = x_in + ((size_t)(b * N + n) * T) * 3;
  const float* vp = v_in + ((size_t)(b * N + n) * T) * 3;
  float ang[T];
  for (int t = 0; t < T; t++) {
    int tp = (t == 0) ? 0 : t - 1;
    float dp = 0, n1 = 0, n2 = 0;
    for (int d = 0; d < 3; d++) {
      float a = vp[tp * 3 + d], c = vp[t * 3 + d];
      dp += a * c; n1 += a * a; n2 += c * c;
    }
    float cv = dp / ((sqrtf(n1) + EPSF) * (sqrtf(n2) + EPSF));
    cv = fminf(1.0f, fmaxf(-1.0f, cv));
    ang[t] = acosf(cv);
  }
  float* hhp = hh + (size_t)(b * N + n) * H;
  for (int j = 0; j < H / 2; j++) {
    float s1 = emb_b[j], s2 = emb2_b[j];
    for (int t = 0; t < T; t++) {
      s1 += h_in[(size_t)(b * N + n) * T + t] * emb_w[t * (H / 2) + j];
      s2 += ang[t] * emb2_w[t * (H / 2) + j];
    }
    hhp[j] = s1;
    hhp[H / 2 + j] = s2;
  }
  float xcv[3] = {xcen[b * 3 + 0], xcen[b * 3 + 1], xcen[b * 3 + 2]};
  float xmv[3] = {xmean[b * 3 + 0], xmean[b * 3 + 1], xmean[b * 3 + 2]};
  float xt[T][3], vt[T][3];
  for (int k = 0; k < T; k++) {
    float wk = (k == 0) ? sqrtf(1.0f / T) : sqrtf(2.0f / T);
    for (int d = 0; d < 3; d++) { xt[k][d] = 0; vt[k][d] = 0; }
    for (int t = 0; t < T; t++) {
      float cw = wk * cosf(PI_F * (t + 0.5f) * k / T);
      for (int d = 0; d < 3; d++) {
        xt[k][d] += cw * (xp[t * 3 + d] - xcv[d]);
        vt[k][d] += cw * vp[t * 3 + d];
      }
    }
  }
  for (int c = 0; c < C; c++) {
    for (int d = 0; d < 3; d++) {
      float sx = 0, sv = 0;
      for (int k = 0; k < T; k++) {
        sx += (xt[k][d] - xmv[d]) * ctw[k * C + c];
        sv += vt[k][d] * vtw[k * C + c];
      }
      xc[((size_t)(b * N + n) * C + c) * 3 + d] = sx + xmv[d];
      vc[((size_t)(b * N + n) * C + c) * 3 + d] = sv;
    }
  }
}

// ---------------------------------------------------------------------------
// calc_category edge phase: dist -> cat_c MLP -> dist' (stored f16) and
// cat_e MLP -> ef, masked-summed per node. One block per (b,m), 4 waves x 2
// n-tiles of 16 edges. All matmuls via WMMA.
// ---------------------------------------------------------------------------
struct CatW {
  const h16 *c1T, *c2T, *e1T, *e2T;
  const float *c_b1, *c_b2, *e_b1, *e_b2;
};

__global__ __launch_bounds__(128) void k_cat_edges(
    const float* __restrict__ hh, const float* __restrict__ xc,
    const float* __restrict__ vc, CatW w, h16* __restrict__ distp,
    float* __restrict__ sumef) {
  constexpr int LDA = 264, LDH = 72;
  __shared__ __attribute__((aligned(16))) h16 bufIn[NW][16 * LDA];
  __shared__ __attribute__((aligned(16))) h16 bufHid[NW][16 * LDH];
  __shared__ float coordm[TWOC * 3];
  __shared__ float hhm[H];
  __shared__ float sred[NW][H];
  int b = blockIdx.x / N, m = blockIdx.x % N;
  int wid = threadIdx.x >> 5, lane = threadIdx.x & 31;
  for (int i = threadIdx.x; i < TWOC * 3; i += blockDim.x) {
    int ch = i / 3, d = i % 3;
    coordm[i] = (ch < C) ? xc[((size_t)(b * N + m) * C + ch) * 3 + d]
                         : vc[((size_t)(b * N + m) * C + (ch - C)) * 3 + d];
  }
  for (int i = threadIdx.x; i < H; i += blockDim.x) hhm[i] = hh[(size_t)(b * N + m) * H + i];
  for (int i = lane; i < H; i += 32) sred[wid][i] = 0.0f;
  __syncthreads();
  h16* bi = bufIn[wid];
  h16* bh = bufHid[wid];
  for (int tl = wid; tl < NTILE; tl += NW) {
    int n0 = tl * 16;
    // pairwise distances (coords ++ velocity channels) into cols 0..127
    for (int idx = lane; idx < 16 * TWOC; idx += 32) {
      int e = idx >> 7, ch = idx & 127;
      int n = n0 + e;
      const float* cn = (ch < C) ? (xc + ((size_t)(b * N + n) * C + ch) * 3)
                                 : (vc + ((size_t)(b * N + n) * C + (ch - C)) * 3);
      float s = 1e-12f;
      for (int d = 0; d < 3; d++) { float df = coordm[ch * 3 + d] - cn[d]; s += df * df; }
      bi[e * LDA + ch] = (h16)sqrtf(s);
    }
    __syncthreads();
    // cat_c: 128 -> 64 (silu)
    gemm16<128, 64>(bi, LDA, w.c1T, [&](int, int, int row, int col, float v) {
      bh[row * LDH + col] = (h16)siluf(v + w.c_b1[col]);
    });
    __syncthreads();
    // cat_c: 64 -> 128 (silu) -> dist' into cols 128..255 + global
    gemm16<64, 128>(bh, LDH, w.c2T, [&](int, int, int row, int col, float v) {
      h16 ho = (h16)siluf(v + w.c_b2[col]);
      bi[row * LDA + 128 + col] = ho;
      distp[((size_t)(b * N + m) * N + (n0 + row)) * TWOC + col] = ho;
    });
    __syncthreads();
    // concat [h1 | h2] into cols 0..127
    for (int idx = lane; idx < 16 * TWOC; idx += 32) {
      int e = idx >> 7, col = idx & 127;
      float v = (col < H) ? hhm[col] : hh[(size_t)(b * N + (n0 + e)) * H + (col - H)];
      bi[e * LDA + col] = (h16)v;
    }
    __syncthreads();
    // cat_e: 256 -> 64 (silu)
    gemm16<256, 64>(bi, LDA, w.e1T, [&](int, int, int row, int col, float v) {
      bh[row * LDH + col] = (h16)siluf(v + w.e_b1[col]);
    });
    __syncthreads();
    // cat_e: 64 -> 64 (silu), in-place
    gemm16<64, 64>(bh, LDH, w.e2T, [&](int, int, int row, int col, float v) {
      bh[row * LDH + col] = (h16)siluf(v + w.e_b2[col]);
    });
    __syncthreads();
    // masked sum over edges (exclude self-edge)
    int ediag = m - n0;
    for (int c = lane; c < H; c += 32) {
      float s = 0;
      for (int e = 0; e < 16; e++)
        if (e != ediag) s += (float)bh[e * LDH + c];
      sred[wid][c] += s;
    }
    __syncthreads();
  }
  __syncthreads();
  for (int c = threadIdx.x; c < H; c += blockDim.x) {
    float s = 0;
    for (int wv = 0; wv < NW; wv++) s += sred[wv][c];
    sumef[(size_t)(b * N + m) * H + c] = s;
  }
}

// node = mlp2(concat[h, sum_ef]) — one block per node, VALU
__global__ __launch_bounds__(64) void k_node_mlp(
    const float* __restrict__ hh, const float* __restrict__ sumef,
    const float* __restrict__ w1, const float* __restrict__ b1,
    const float* __restrict__ w2, const float* __restrict__ b2,
    float* __restrict__ out) {
  __shared__ float in[2 * H];
  __shared__ float hid[H];
  int id = blockIdx.x, t = threadIdx.x;
  in[t] = hh[(size_t)id * H + t];
  in[H + t] = sumef[(size_t)id * H + t];
  __syncthreads();
  float s = b1[t];
  for (int i = 0; i < 2 * H; i++) s += in[i] * w1[i * H + t];
  hid[t] = siluf(s);
  __syncthreads();
  float s2 = b2[t];
  for (int i = 0; i < H; i++) s2 += hid[i] * w2[i * H + t];
  out[(size_t)id * H + t] = siluf(s2);
}

// logits + softmax -> category (written straight into d_out region)
__global__ __launch_bounds__(128) void k_logits(
    const float* __restrict__ node, const h16* __restrict__ distp,
    const h16* __restrict__ k1T, const float* __restrict__ k_b1,
    const float* __restrict__ k_w2, const float* __restrict__ k_b2,
    float* __restrict__ catOut) {
  constexpr int LDA = 264, LDH = 72;
  __shared__ __attribute__((aligned(16))) h16 bufIn[NW][16 * LDA];
  __shared__ __attribute__((aligned(16))) h16 bufHid[NW][16 * LDH];
  __shared__ float nodem[H];
  int b = blockIdx.x / N, m = blockIdx.x % N;
  int wid = threadIdx.x >> 5, lane = threadIdx.x & 31;
  for (int i = threadIdx.x; i < H; i += blockDim.x) nodem[i] = node[(size_t)(b * N + m) * H + i];
  __syncthreads();
  h16* bi = bufIn[wid];
  h16* bh = bufHid[wid];
  for (int tl = wid; tl < NTILE; tl += NW) {
    int n0 = tl * 16;
    for (int idx = lane; idx < 16 * 256; idx += 32) {
      int e = idx >> 8, col = idx & 255;
      int n = n0 + e;
      h16 v;
      if (col < H) v = (h16)nodem[col];
      else if (col < 2 * H) v = (h16)node[(size_t)(b * N + n) * H + (col - H)];
      else v = distp[((size_t)(b * N + m) * N + n) * TWOC + (col - 2 * H)];
      bi[e * LDA + col] = v;
    }
    __syncthreads();
    gemm16<256, 64>(bi, LDA, k1T, [&](int, int, int row, int col, float v) {
      bh[row * LDH + col] = (h16)siluf(v + k_b1[col]);
    });
    __syncthreads();
    {  // final 64 -> K=2 + softmax(logits/TAO); lane = edge*2 + k
      int e = lane >> 1, k = lane & 1;
      float s = k_b2[k];
      for (int g = 0; g < H; g++) s += (float)bh[e * LDH + g] * k_w2[g * KCAT + k];
      float z = s / TAO;
      float other = __shfl_xor(z, 1, 32);
      float mx = fmaxf(z, other);
      float e0 = __expf(z - mx), e1 = __expf(other - mx);
      catOut[((size_t)(b * N + m) * N + (n0 + e)) * KCAT + k] = e0 / (e0 + e1);
    }
    __syncthreads();
  }
}

// ---------------------------------------------------------------------------
// gcl layer edge phase: radial -> e-MLP -> ef; per-category channel MLPs mixed
// by inferred category; equivariant aggregation over neighbors.
// ---------------------------------------------------------------------------
struct LayW {
  const h16 *e1T, *e2T, *c1T, *c2T;  // c1T/c2T: [KCAT][64*64] transposed
  const float *e_b1, *e_b2, *c_b1;   // c_b1: [KCAT][64]
};

__global__ __launch_bounds__(128) void k_layer_edges(
    const float* __restrict__ hh, const float* __restrict__ xc,
    const float* __restrict__ category, LayW w, float* __restrict__ agg,
    float* __restrict__ sumef) {
  constexpr int LDA = 200, LDH = 72;
  __shared__ __attribute__((aligned(16))) h16 bufIn[NW][16 * LDA];
  __shared__ __attribute__((aligned(16))) h16 bufE[NW][16 * LDH];
  __shared__ __attribute__((aligned(16))) h16 bufT[NW][16 * LDH];
  __shared__ h16 wbuf[NW][16 * C];
  __shared__ float catv[NW][16 * KCAT];
  __shared__ float xm[C * 3];
  __shared__ float hhm[H];
  __shared__ float sred[NW][H];
  __shared__ float ared[NW][C * 3];
  int b = blockIdx.x / N, m = blockIdx.x % N;
  int wid = threadIdx.x >> 5, lane = threadIdx.x & 31;
  for (int i = threadIdx.x; i < C * 3; i += blockDim.x)
    xm[i] = xc[((size_t)(b * N + m) * C) * 3 + i];
  for (int i = threadIdx.x; i < H; i += blockDim.x) hhm[i] = hh[(size_t)(b * N + m) * H + i];
  for (int i = lane; i < H; i += 32) sred[wid][i] = 0.0f;
  for (int i = lane; i < C * 3; i += 32) ared[wid][i] = 0.0f;
  __syncthreads();
  h16* bi = bufIn[wid];
  h16* be = bufE[wid];
  h16* bt = bufT[wid];
  h16* wb = wbuf[wid];
  for (int tl = wid; tl < NTILE; tl += NW) {
    int n0 = tl * 16;
    {  // per-edge category weights for this tile
      int e = lane >> 1, k = lane & 1;
      catv[wid][e * KCAT + k] = category[((size_t)(b * N + m) * N + (n0 + e)) * KCAT + k];
    }
    // concat [h1 | h2 | radial]
    for (int idx = lane; idx < 16 * (2 * H + C); idx += 32) {
      int e = idx / 192, col = idx % 192;
      int n = n0 + e;
      float v;
      if (col < H) v = hhm[col];
      else if (col < 2 * H) v = hh[(size_t)(b * N + n) * H + (col - H)];
      else {
        int c = col - 2 * H;
        const float* xn = xc + ((size_t)(b * N + n) * C + c) * 3;
        float s = 0;
        for (int d = 0; d < 3; d++) { float df = xm[c * 3 + d] - xn[d]; s += df * df; }
        v = s;
      }
      bi[e * LDA + col] = (h16)v;
    }
    __syncthreads();
    gemm16<192, 64>(bi, LDA, w.e1T, [&](int, int, int row, int col, float v) {
      be[row * LDH + col] = (h16)siluf(v + w.e_b1[col]);
    });
    __syncthreads();
    gemm16<64, 64>(be, LDH, w.e2T, [&](int, int, int row, int col, float v) {
      be[row * LDH + col] = (h16)siluf(v + w.e_b2[col]);
    });
    __syncthreads();
    int ediag = m - n0;
    for (int c = lane; c < H; c += 32) {
      float s = 0;
      for (int e = 0; e < 16; e++)
        if (e != ediag) s += (float)be[e * LDH + c];
      sred[wid][c] += s;
    }
    // category-mixed equivariant channel weights w[e][c]
    for (int k = 0; k < KCAT; k++) {
      gemm16<64, 64>(be, LDH, w.c1T + (size_t)k * H * H,
                     [&](int, int, int row, int col, float v) {
                       bt[row * LDH + col] = (h16)siluf(v + w.c_b1[k * H + col]);
                     });
      __syncthreads();
      if (k == 0) {
        gemm16<64, 64>(bt, LDH, w.c2T + (size_t)k * H * C,
                       [&](int, int, int row, int col, float v) {
                         wb[row * C + col] = (h16)(catv[wid][row * KCAT + k] * v);
                       });
      } else {
        gemm16<64, 64>(bt, LDH, w.c2T + (size_t)k * H * C,
                       [&](int, int, int row, int col, float v) {
                         wb[row * C + col] =
                             (h16)((float)wb[row * C + col] + catv[wid][row * KCAT + k] * v);
                       });
      }
      __syncthreads();
    }
    // equivariant aggregation: agg[c][d] += sum_e w[e][c] * (xm - xn)[c][d]
    for (int idx = lane; idx < C * 3; idx += 32) {
      int c = idx / 3, d = idx % 3;
      float s = 0;
      for (int e = 0; e < 16; e++) {
        if (e == ediag) continue;
        int n = n0 + e;
        float wv = (float)wb[e * C + c];
        float xn = xc[((size_t)(b * N + n) * C + c) * 3 + d];
        s += wv * (xm[idx] - xn);
      }
      ared[wid][idx] += s;
    }
    __syncthreads();
  }
  __syncthreads();
  for (int i = threadIdx.x; i < H; i += blockDim.x) {
    float s = 0;
    for (int wv = 0; wv < NW; wv++) s += sred[wv][i];
    sumef[(size_t)(b * N + m) * H + i] = s;
  }
  for (int i = threadIdx.x; i < C * 3; i += blockDim.x) {
    float s = 0;
    for (int wv = 0; wv < NW; wv++) s += ared[wv][i];
    agg[((size_t)(b * N + m) * C) * 3 + i] = s * (1.0f / (N - 1));
  }
}

// gcl layer node phase: x += agg + (h @ v_w + v_b) * vel ; h = mlp2([h, sum_ef])
__global__ __launch_bounds__(64) void k_layer_nodes(
    const float* __restrict__ agg, const float* __restrict__ sumef,
    const float* __restrict__ vc, const float* __restrict__ v_w,
    const float* __restrict__ v_b, const float* __restrict__ n_w1,
    const float* __restrict__ n_b1, const float* __restrict__ n_w2,
    const float* __restrict__ n_b2, float* __restrict__ hh, float* __restrict__ xc) {
  __shared__ float in[2 * H];
  __shared__ float hid[H];
  int id = blockIdx.x, t = threadIdx.x;
  float coef = v_b[t];
  for (int i = 0; i < H; i++) coef += hh[(size_t)id * H + i] * v_w[i * C + t];
  for (int d = 0; d < 3; d++) {
    size_t xi = ((size_t)id * C + t) * 3 + d;
    xc[xi] = xc[xi] + agg[xi] + coef * vc[xi];
  }
  in[t] = hh[(size_t)id * H + t];
  in[H + t] = sumef[(size_t)id * H + t];
  __syncthreads();
  float s = n_b1[t];
  for (int i = 0; i < 2 * H; i++) s += in[i] * n_w1[i * H + t];
  hid[t] = siluf(s);
  __syncthreads();
  float s2 = n_b2[t];
  for (int i = 0; i < H; i++) s2 += hid[i] * n_w2[i * H + t];
  float newh = siluf(s2);
  __syncthreads();
  hh[(size_t)id * H + t] = newh;
}

// mean over (n,c) of xc per batch/dim
__global__ void k_mean_xc(const float* __restrict__ xc, float* __restrict__ xm2) {
  __shared__ float red[256];
  int b = blockIdx.x / 3, d = blockIdx.x % 3;
  float s = 0;
  for (int i = threadIdx.x; i < N * C; i += 256)
    s += xc[(((size_t)b * N * C) + i) * 3 + d];
  red[threadIdx.x] = s;
  __syncthreads();
  for (int st = 128; st > 0; st >>= 1) {
    if (threadIdx.x < st) red[threadIdx.x] += red[threadIdx.x + st];
    __syncthreads();
  }
  if (threadIdx.x == 0) xm2[b * 3 + d] = red[0] / (float)(N * C);
}

// predict head + IDCT (orthonormal DCT inverse == transpose) + recenter
__global__ __launch_bounds__(64) void k_predict(
    const float* __restrict__ xc, const float* __restrict__ pw,
    const float* __restrict__ xm2, const float* __restrict__ xcen,
    float* __restrict__ xout) {
  __shared__ float xo[OUTT * 3];
  int b = blockIdx.x / N, n = blockIdx.x % N;
  int t = threadIdx.x;
  if (t < OUTT * 3) {
    int o = t / 3, d = t % 3;
    float xm = xm2[b * 3 + d];
    float s = 0;
    for (int c = 0; c < C; c++)
      s += (xc[((size_t)(b * N + n) * C + c) * 3 + d] - xm) * pw[c * OUTT + o];
    xo[t] = s + xm;
  }
  __syncthreads();
  if (t < OUTT * 3) {
    int k = t / 3, d = t % 3;
    float s = xcen[b * 3 + d];
    for (int o = 0; o < OUTT; o++) {
      float wo = (o == 0) ? sqrtf(1.0f / OUTT) : sqrtf(2.0f / OUTT);
      float idct = wo * cosf(PI_F * (k + 0.5f) * o / OUTT);
      s += idct * xo[o * 3 + d];
    }
    xout[((size_t)(b * N + n) * OUTT + k) * 3 + d] = s;
  }
}

// ---------------------------------------------------------------------------
extern "C" void kernel_launch(void* const* d_in, const int* in_sizes, int n_in,
                              void* d_out, int out_size, void* d_ws, size_t ws_size,
                              hipStream_t stream) {
  (void)in_sizes; (void)n_in; (void)out_size; (void)ws_size;
  auto inf = [&](int i) { return (const float*)d_in[i]; };
  // d_in layout (setup_inputs dict order, params flattened in insertion order)
  const float* h_in = inf(0);
  const float* x_in = inf(1);
  const float* v_in = inf(2);
  // 3..9: emb_w, emb_b, emb2_w, emb2_b, coord_trans_w, vel_trans_w, predict_w
  // 10..25: cat_c_w1/b1/w2/b2, cat_e_*, cat_n_*, cat_k_*
  // 26 + 13*l: e_w1,e_b1,e_w2,e_b2,c_w1,c_b1,c_w2,v_w,v_b,n_w1,n_b1,n_w2,n_b2

  // -------- workspace layout --------
  float* fw = (float*)d_ws;
  float* hh = fw;     fw += (size_t)B * N * H;
  float* xcb = fw;    fw += (size_t)B * N * C * 3;
  float* vcb = fw;    fw += (size_t)B * N * C * 3;
  float* sumef = fw;  fw += (size_t)B * N * H;
  float* node = fw;   fw += (size_t)B * N * H;
  float* aggb = fw;   fw += (size_t)B * N * C * 3;
  float* Mb = fw;     fw += B * T * 3;
  float* xcen = fw;   fw += B * 3;
  float* xmean = fw;  fw += B * 3;
  float* xm2 = fw;    fw += B * 3;
  h16* hb = (h16*)(((uintptr_t)fw + 31) & ~(uintptr_t)31);
  h16* distp = hb;    hb += (size_t)B * N * N * TWOC;
  h16* cc1T = hb;     hb += 64 * 128;
  h16* cc2T = hb;     hb += 128 * 64;
  h16* ce1T = hb;     hb += 64 * 256;
  h16* ce2T = hb;     hb += 64 * 64;
  h16* ck1T = hb;     hb += 64 * 256;
  h16* le1T[L]; h16* le2T[L]; h16* lc1T[L]; h16* lc2T[L];
  for (int l = 0; l < L; l++) {
    le1T[l] = hb; hb += 64 * 192;
    le2T[l] = hb; hb += 64 * 64;
    lc1T[l] = hb; hb += KCAT * 64 * 64;
    lc2T[l] = hb; hb += KCAT * 64 * 64;
  }

  float* xout = (float*)d_out;
  float* catOut = (float*)d_out + (size_t)B * N * OUTT * 3;

  // -------- weight conversion (f32 [in][out] -> f16 transposed [out][in]) --------
  auto tr = [&](const float* src, h16* dst, int in, int out) {
    int n = in * out;
    k_transpose_f16<<<(n + 255) / 256, 256, 0, stream>>>(src, dst, in, out);
  };
  tr(inf(10), cc1T, 128, 64);
  tr(inf(12), cc2T, 64, 128);
  tr(inf(14), ce1T, 256, 64);
  tr(inf(16), ce2T, 64, 64);
  tr(inf(22), ck1T, 256, 64);
  for (int l = 0; l < L; l++) {
    int base = 26 + 13 * l;
    tr(inf(base + 0), le1T[l], 192, 64);
    tr(inf(base + 2), le2T[l], 64, 64);
    tr(inf(base + 4), lc1T[l], 64, 64);
    tr(inf(base + 4) + 64 * 64, lc1T[l] + 64 * 64, 64, 64);
    tr(inf(base + 6), lc2T[l], 64, 64);
    tr(inf(base + 6) + 64 * 64, lc2T[l] + 64 * 64, 64, 64);
  }

  // -------- node prep --------
  k_mean_nt<<<B * T * 3, 128, 0, stream>>>(x_in, Mb);
  k_centers<<<1, 32, 0, stream>>>(Mb, xcen, xmean);
  k_prep<<<B, N, 0, stream>>>(h_in, x_in, v_in, inf(3), inf(4), inf(5), inf(6),
                              inf(7), inf(8), xcen, xmean, hh, xcb, vcb);

  // -------- category --------
  CatW cw;
  cw.c1T = cc1T; cw.c2T = cc2T; cw.e1T = ce1T; cw.e2T = ce2T;
  cw.c_b1 = inf(11); cw.c_b2 = inf(13); cw.e_b1 = inf(15); cw.e_b2 = inf(17);
  k_cat_edges<<<B * N, 128, 0, stream>>>(hh, xcb, vcb, cw, distp, sumef);
  k_node_mlp<<<B * N, 64, 0, stream>>>(hh, sumef, inf(18), inf(19), inf(20), inf(21), node);
  k_logits<<<B * N, 128, 0, stream>>>(node, distp, ck1T, inf(23), inf(24), inf(25), catOut);

  // -------- interaction layers --------
  for (int l = 0; l < L; l++) {
    int base = 26 + 13 * l;
    LayW lw;
    lw.e1T = le1T[l]; lw.e2T = le2T[l]; lw.c1T = lc1T[l]; lw.c2T = lc2T[l];
    lw.e_b1 = inf(base + 1); lw.e_b2 = inf(base + 3); lw.c_b1 = inf(base + 5);
    k_layer_edges<<<B * N, 128, 0, stream>>>(hh, xcb, catOut, lw, aggb, sumef);
    k_layer_nodes<<<B * N, 64, 0, stream>>>(aggb, sumef, vcb, inf(base + 7),
                                            inf(base + 8), inf(base + 9), inf(base + 10),
                                            inf(base + 11), inf(base + 12), hh, xcb);
  }

  // -------- predict head --------
  k_mean_xc<<<B * 3, 256, 0, stream>>>(xcb, xm2);
  k_predict<<<B * N, 64, 0, stream>>>(xcb, inf(9), xm2, xcen, xout);
}
